// TransformerOCModel_30296699306419
// MI455X (gfx1250) — matrix operand order repaired and checked by reading, not compiled, not run on hardware
//
#include <hip/hip_runtime.h>
#include <hip/hip_bf16.h>

// -------- model constants (fixed by the reference) --------
#define NTOK   4096
#define INDIM  16
#define HDIM   128
#define NLAYER 4
#define NHEAD  8
#define HEADD  16
#define FFDIM  256
#define QKVDIM 384
#define MASKW  (NTOK / 32)

// -------- WMMA fragment types (CDNA5 gfx1250, wave32) --------
typedef __attribute__((ext_vector_type(16))) __bf16 v16bf;
typedef __attribute__((ext_vector_type(8)))  float  v8f;

struct __align__(16) U4 { unsigned int x, y, z, w; };

union FragAB {
    v16bf v;
    U4 q[2];
    unsigned short h[16];
};
union FragC {
    v8f v;
    float e[8];
};

__device__ __forceinline__ unsigned short f2bf(float f) {
    unsigned int u = __float_as_uint(f);
    unsigned int r = u + 0x7FFFu + ((u >> 16) & 1u);   // round-to-nearest-even
    return (unsigned short)(r >> 16);
}
__device__ __forceinline__ float bf2f(unsigned short h) {
    return __uint_as_float(((unsigned int)h) << 16);
}
__device__ __forceinline__ U4 u4zero() { U4 z; z.x = z.y = z.z = z.w = 0u; return z; }

// =====================================================================
// f32 -> bf16 elementwise convert (weights + activations)
// =====================================================================
__global__ void cvt_bf16_kernel(const float* __restrict__ in,
                                unsigned short* __restrict__ out, int n) {
    int i = blockIdx.x * blockDim.x + threadIdx.x;
    if (i < n) out[i] = f2bf(in[i]);
}

// =====================================================================
// eta/phi extraction + packed ΔR bitmask (1 bit per pair, bit=1: allowed)
// =====================================================================
__global__ void etaphi_kernel(const float* __restrict__ x_raw, float* __restrict__ ep) {
    int i = blockIdx.x * blockDim.x + threadIdx.x;
    if (i >= NTOK) return;
    ep[i * 2 + 0] = x_raw[i * INDIM + 1] * 5.24f - 2.62f;      // eta
    ep[i * 2 + 1] = x_raw[i * INDIM + 2] * 6.2832f - 3.1416f;  // phi
}

__global__ void maskbits_kernel(const float* __restrict__ ep, unsigned int* __restrict__ mb) {
    int idx = blockIdx.x * blockDim.x + threadIdx.x;   // NTOK * MASKW threads
    if (idx >= NTOK * MASKW) return;
    int i = idx / MASKW;
    int w = idx % MASKW;
    float ei = ep[i * 2 + 0], pi_ = ep[i * 2 + 1];
    unsigned int bits = 0u;
    int j0 = w * 32;
    #pragma unroll 8
    for (int t = 0; t < 32; t++) {
        int j = j0 + t;
        float de = ei - ep[j * 2 + 0];
        float dp = pi_ - ep[j * 2 + 1];
        dp = dp - 6.2832f * rintf(dp * (1.0f / 6.2832f));  // wrap to [-pi, pi]
        float dr2 = de * de + dp * dp;
        if (dr2 <= 0.04f) bits |= (1u << t);               // DR_THR^2 = 0.04
    }
    mb[idx] = bits;
}

// =====================================================================
// input embedding: x = x_raw @ in_w^T + in_b  (tiny K=16, plain f32)
// =====================================================================
__global__ void embed_kernel(const float* __restrict__ xr, const float* __restrict__ w,
                             const float* __restrict__ b, float* __restrict__ x) {
    int idx = blockIdx.x * blockDim.x + threadIdx.x;
    if (idx >= NTOK * HDIM) return;
    int row = idx / HDIM, c = idx % HDIM;
    float acc = b[c];
    #pragma unroll
    for (int k = 0; k < INDIM; k++) acc += xr[row * INDIM + k] * w[c * INDIM + k];
    x[idx] = acc;
}

// =====================================================================
// LayerNorm over H=128, one 128-thread block per row, bf16 output
// =====================================================================
__global__ void ln_bf16_kernel(const float* __restrict__ x, const float* __restrict__ s,
                               const float* __restrict__ b, unsigned short* __restrict__ out) {
    __shared__ float red[HDIM];
    __shared__ float mv[2];
    int row = blockIdx.x, t = threadIdx.x;
    float v = x[(size_t)row * HDIM + t];
    red[t] = v; __syncthreads();
    for (int off = HDIM / 2; off > 0; off >>= 1) {
        if (t < off) red[t] += red[t + off];
        __syncthreads();
    }
    if (t == 0) mv[0] = red[0] * (1.0f / HDIM);
    __syncthreads();
    float d = v - mv[0];
    red[t] = d * d; __syncthreads();
    for (int off = HDIM / 2; off > 0; off >>= 1) {
        if (t < off) red[t] += red[t + off];
        __syncthreads();
    }
    if (t == 0) mv[1] = rsqrtf(red[0] * (1.0f / HDIM) + 1e-5f);
    __syncthreads();
    out[(size_t)row * HDIM + t] = f2bf(d * mv[1] * s[t] + b[t]);
}

// =====================================================================
// Generic bf16 WMMA GEMM: C[M,Nc] = A[M,K] @ W[Nc,K]^T (+bias)(+relu)(+resid)
// One wave computes one 16x16 tile with v_wmma_f32_16x16x32_bf16 over K.
// A fragment:  lane holds row m0+lane%16; v0..3 = K[kk + (lane/16)*8 ..+7],
//              v4..7 = K[kk+16 + (lane/16)*8 ..+7]  (contiguous b128 loads)
// B fragment:  lane holds col n0+lane%16; v0..3 = K[kk + (lane/16)*8 ..+7],
//              v4..7 = K[kk+16 + (lane/16)*8 ..+7] (two b128 loads from W row)
// C layout:    lane = col n0+lane%16, vgpr r = row m0 + r + 8*(lane/16)
// =====================================================================
__global__ void gemm_bf16_wmma(const unsigned short* __restrict__ A,
                               const unsigned short* __restrict__ W,
                               const float* __restrict__ bias,
                               const float* __restrict__ resid,
                               float* __restrict__ Cf,
                               unsigned short* __restrict__ Cb,
                               int M, int K, int Nc, int relu, int nwaves) {
    int gtid = blockIdx.x * blockDim.x + threadIdx.x;
    int wave = gtid >> 5;
    int lane = gtid & 31;
    if (wave >= nwaves) return;
    int tilesN = Nc >> 4;
    int m0 = (wave / tilesN) << 4;
    int n0 = (wave % tilesN) << 4;
    int col16 = lane & 15;
    int lh    = lane >> 4;

    const unsigned short* arow = A + (size_t)(m0 + col16) * K + lh * 8;
    const unsigned short* wrow = W + (size_t)(n0 + col16) * K + lh * 8;

    FragC c;
    #pragma unroll
    for (int r = 0; r < 8; r++) c.e[r] = 0.0f;

    for (int kk = 0; kk < K; kk += 32) {
        FragAB a, b;
        a.q[0] = *reinterpret_cast<const U4*>(arow + kk);
        a.q[1] = *reinterpret_cast<const U4*>(arow + kk + 16);
        b.q[0] = *reinterpret_cast<const U4*>(wrow + kk);
        b.q[1] = *reinterpret_cast<const U4*>(wrow + kk + 16);
        c.v = __builtin_amdgcn_wmma_f32_16x16x32_bf16(false, a.v, false, b.v,
                                                      (short)0, c.v, false, false);
    }

    int col = n0 + col16;
    float bv = bias ? bias[col] : 0.0f;
    #pragma unroll
    for (int r = 0; r < 8; r++) {
        int row = m0 + r + lh * 8;
        float v = c.e[r] + bv;
        if (relu) v = v > 0.0f ? v : 0.0f;
        size_t off = (size_t)row * Nc + col;
        if (resid) v += resid[off];
        if (Cf) Cf[off] = v;
        if (Cb) Cb[off] = f2bf(v);
    }
}

// =====================================================================
// Flash attention, one wave per (head, 16-query block).
// QKV is [NTOK][384] bf16 (cols: Q 0..127 | K 128..255 | V 256..383,
// head h occupying 16 contiguous cols). Head-dim 16 is zero-padded to
// WMMA K=32. S tile masked via packed dr-bitmask; online softmax with
// __shfl_xor reductions inside 16-lane halves; P and V^T staged through
// LDS to re-layout for the P@V WMMA.
// =====================================================================
__global__ __launch_bounds__(256)
void attn_wmma(const unsigned short* __restrict__ QKV,
               const unsigned int* __restrict__ mb,
               unsigned short* __restrict__ O) {
    __shared__ unsigned short lds_p[8][256];
    __shared__ unsigned short lds_vt[8][256];
    int wslot = threadIdx.x >> 5;
    int lane  = threadIdx.x & 31;
    int gwave = blockIdx.x * 8 + wslot;
    int h  = gwave & (NHEAD - 1);
    int qb = gwave >> 3;                 // 0..255, exact grid
    int q0 = qb << 4;
    int col16 = lane & 15;
    int lh    = lane >> 4;

    // Q fragment (A): row q0+col16, K dims 0..15 real, 16..31 zero pad
    FragAB aq;
    aq.q[0] = *reinterpret_cast<const U4*>(QKV + (size_t)(q0 + col16) * QKVDIM + h * HEADD + lh * 8);
    aq.q[1] = u4zero();

    float mrow[8], lrow[8];
    FragC o;
    #pragma unroll
    for (int r = 0; r < 8; r++) { mrow[r] = -1e30f; lrow[r] = 0.0f; o.e[r] = 0.0f; }

    for (int kb = 0; kb < NTOK / 16; kb++) {
        int k0 = kb << 4;
        // K fragment (B): B(d,j) = K[k0+j][d]; elements 0..7 carry d=(lane/16)*8..+7,
        // elements 8..15 carry the zero pad.
        FragAB bk;
        {
            const unsigned short* kr = QKV + (size_t)(k0 + col16) * QKVDIM + HDIM + h * HEADD + lh * 8;
            bk.q[0] = *reinterpret_cast<const U4*>(kr);
            bk.q[1] = u4zero();
        }
        FragC s;
        #pragma unroll
        for (int r = 0; r < 8; r++) s.e[r] = 0.0f;
        s.v = __builtin_amdgcn_wmma_f32_16x16x32_bf16(false, aq.v, false, bk.v,
                                                      (short)0, s.v, false, false);

        // mask + online softmax (row = (r, lane-half); cols across 16 lanes)
        unsigned int wordIdx = (unsigned)(k0 >> 5);
        int bitBase = (k0 & 31) + col16;
        #pragma unroll
        for (int r = 0; r < 8; r++) {
            int q = q0 + r + lh * 8;
            unsigned int w = mb[(size_t)q * MASKW + wordIdx];
            float sv = ((w >> bitBase) & 1u) ? s.e[r] * 0.25f : -1e30f; // scale=1/sqrt(16)
            float rmax = sv;
            rmax = fmaxf(rmax, __shfl_xor(rmax, 1, 32));
            rmax = fmaxf(rmax, __shfl_xor(rmax, 2, 32));
            rmax = fmaxf(rmax, __shfl_xor(rmax, 4, 32));
            rmax = fmaxf(rmax, __shfl_xor(rmax, 8, 32));
            float mnew  = fmaxf(mrow[r], rmax);
            float alpha = __expf(mrow[r] - mnew);
            float p     = __expf(sv - mnew);
            float psum  = p;
            psum += __shfl_xor(psum, 1, 32);
            psum += __shfl_xor(psum, 2, 32);
            psum += __shfl_xor(psum, 4, 32);
            psum += __shfl_xor(psum, 8, 32);
            lrow[r] = lrow[r] * alpha + psum;
            o.e[r] *= alpha;
            mrow[r] = mnew;
            lds_p[wslot][(r + lh * 8) * 16 + col16] = f2bf(p);   // P[q_local][j]
        }
        // stage V transposed: lds_vt[d][j] = V[k0+j][d]
        {
            const unsigned short* vr = QKV + (size_t)(k0 + col16) * QKVDIM + 2 * HDIM + h * HEADD + lh * 8;
            U4 vq = *reinterpret_cast<const U4*>(vr);
            union { U4 q; unsigned short hh[8]; } vu; vu.q = vq;
            #pragma unroll
            for (int t = 0; t < 8; t++)
                lds_vt[wslot][(lh * 8 + t) * 16 + col16] = vu.hh[t];
        }
        __syncthreads();

        // P@V: A = P (rows q, K = key index j, padded to 32), B = V^T
        FragAB ap, bvf;
        ap.q[0] = *reinterpret_cast<const U4*>(&lds_p[wslot][col16 * 16 + lh * 8]);
        ap.q[1] = u4zero();
        bvf.q[0] = *reinterpret_cast<const U4*>(&lds_vt[wslot][col16 * 16 + lh * 8]);
        bvf.q[1] = u4zero();
        o.v = __builtin_amdgcn_wmma_f32_16x16x32_bf16(false, ap.v, false, bvf.v,
                                                      (short)0, o.v, false, false);
        __syncthreads();
    }

    #pragma unroll
    for (int r = 0; r < 8; r++) {
        int q = q0 + r + lh * 8;
        O[(size_t)q * HDIM + h * HEADD + col16] = f2bf(o.e[r] / lrow[r]);
    }
}

// =====================================================================
// latent L2 normalize (in place on d_out slice), one thread per row
// =====================================================================
__global__ void latnorm_kernel(float* __restrict__ lat) {
    int i = blockIdx.x * blockDim.x + threadIdx.x;
    if (i >= NTOK) return;
    float* p = lat + (size_t)i * 16;
    float ss = 0.0f;
    #pragma unroll
    for (int k = 0; k < 16; k++) ss += p[k] * p[k];
    float inv = 1.0f / fmaxf(sqrtf(ss), 1e-12f);
    #pragma unroll
    for (int k = 0; k < 16; k++) p[k] *= inv;
}

// =====================================================================
// beta head: sigmoid(relu_out[N,64] @ b2_w^T + b2_b), clipped
// =====================================================================
__global__ void beta_kernel(const unsigned short* __restrict__ bb,
                            const float* __restrict__ w, const float* __restrict__ b,
                            float* __restrict__ outb) {
    int i = blockIdx.x * blockDim.x + threadIdx.x;
    if (i >= NTOK) return;
    float acc = b[0];
    #pragma unroll 8
    for (int k = 0; k < 64; k++) acc += bf2f(bb[(size_t)i * 64 + k]) * w[k];
    float sg = 1.0f / (1.0f + __expf(-acc));
    sg = fminf(fmaxf(sg, 1e-6f), 1.0f - 1e-6f);
    outb[i] = sg;
}

// =====================================================================
// host-side orchestration
// =====================================================================
static inline size_t alignup(size_t x) { return (x + 255) & ~(size_t)255; }

extern "C" void kernel_launch(void* const* d_in, const int* in_sizes, int n_in,
                              void* d_out, int out_size, void* d_ws, size_t ws_size,
                              hipStream_t stream) {
    (void)in_sizes; (void)n_in; (void)out_size; (void)ws_size;

    const float* x_raw  = (const float*)d_in[0];
    const float* in_w   = (const float*)d_in[2];
    const float* in_b   = (const float*)d_in[3];
    const float* ln1_s  = (const float*)d_in[4];
    const float* ln1_b  = (const float*)d_in[5];
    const float* qkv_w  = (const float*)d_in[6];
    const float* qkv_b  = (const float*)d_in[7];
    const float* ao_w   = (const float*)d_in[8];
    const float* ao_b   = (const float*)d_in[9];
    const float* ln2_s  = (const float*)d_in[10];
    const float* ln2_b  = (const float*)d_in[11];
    const float* f1_w   = (const float*)d_in[12];
    const float* f1_b   = (const float*)d_in[13];
    const float* f2_w   = (const float*)d_in[14];
    const float* f2_b   = (const float*)d_in[15];
    const float* lat1_w = (const float*)d_in[16];
    const float* lat1_b = (const float*)d_in[17];
    const float* lat2_w = (const float*)d_in[18];
    const float* lat2_b = (const float*)d_in[19];
    const float* b1_w   = (const float*)d_in[20];
    const float* b1_b   = (const float*)d_in[21];
    const float* b2_w   = (const float*)d_in[22];
    const float* b2_b   = (const float*)d_in[23];

    // ---- workspace layout (bump allocator, ~15 MB total) ----
    char* base = (char*)d_ws;
    size_t off = 0;
    auto alloc = [&](size_t bytes) { char* p = base + off; off = alignup(off + bytes); return p; };

    float*          ep      = (float*)         alloc((size_t)NTOK * 2 * 4);
    unsigned int*   mbits   = (unsigned int*)  alloc((size_t)NTOK * MASKW * 4);
    float*          x       = (float*)         alloc((size_t)NTOK * HDIM * 4);
    unsigned short* h_bf    = (unsigned short*)alloc((size_t)NTOK * HDIM * 2);
    unsigned short* qkv_bf  = (unsigned short*)alloc((size_t)NTOK * QKVDIM * 2);
    unsigned short* o_bf    = (unsigned short*)alloc((size_t)NTOK * HDIM * 2);
    unsigned short* ff_bf   = (unsigned short*)alloc((size_t)NTOK * FFDIM * 2);
    unsigned short* xb_bf   = (unsigned short*)alloc((size_t)NTOK * HDIM * 2);
    unsigned short* t_bf    = (unsigned short*)alloc((size_t)NTOK * HDIM * 2);
    unsigned short* bb_bf   = (unsigned short*)alloc((size_t)NTOK * 64 * 2);
    unsigned short* wqkv_bf = (unsigned short*)alloc((size_t)NLAYER * QKVDIM * HDIM * 2);
    unsigned short* wao_bf  = (unsigned short*)alloc((size_t)NLAYER * HDIM * HDIM * 2);
    unsigned short* wf1_bf  = (unsigned short*)alloc((size_t)NLAYER * FFDIM * HDIM * 2);
    unsigned short* wf2_bf  = (unsigned short*)alloc((size_t)NLAYER * HDIM * FFDIM * 2);
    unsigned short* wl1_bf  = (unsigned short*)alloc((size_t)HDIM * HDIM * 2);
    unsigned short* wl2_bf  = (unsigned short*)alloc((size_t)16 * HDIM * 2);
    unsigned short* wb1_bf  = (unsigned short*)alloc((size_t)64 * HDIM * 2);

    float* out_beta = (float*)d_out;
    float* out_lat  = (float*)d_out + NTOK;

    auto cvt = [&](const float* src, unsigned short* dst, int n) {
        cvt_bf16_kernel<<<(n + 255) / 256, 256, 0, stream>>>(src, dst, n);
    };
    auto gemm = [&](const unsigned short* A, const unsigned short* W, const float* bias,
                    const float* resid, float* Cf, unsigned short* Cb,
                    int M, int K, int Nc, int relu) {
        int nwaves = (M / 16) * (Nc / 16);
        int threads = nwaves * 32;
        gemm_bf16_wmma<<<(threads + 255) / 256, 256, 0, stream>>>(
            A, W, bias, resid, Cf, Cb, M, K, Nc, relu, nwaves);
    };

    // ---- one-time (per call) weight conversion to bf16 ----
    cvt(qkv_w,  wqkv_bf, NLAYER * QKVDIM * HDIM);
    cvt(ao_w,   wao_bf,  NLAYER * HDIM * HDIM);
    cvt(f1_w,   wf1_bf,  NLAYER * FFDIM * HDIM);
    cvt(f2_w,   wf2_bf,  NLAYER * HDIM * FFDIM);
    cvt(lat1_w, wl1_bf,  HDIM * HDIM);
    cvt(lat2_w, wl2_bf,  16 * HDIM);
    cvt(b1_w,   wb1_bf,  64 * HDIM);

    // ---- dr mask + embedding ----
    etaphi_kernel<<<(NTOK + 255) / 256, 256, 0, stream>>>(x_raw, ep);
    maskbits_kernel<<<(NTOK * MASKW + 255) / 256, 256, 0, stream>>>(ep, mbits);
    embed_kernel<<<(NTOK * HDIM + 255) / 256, 256, 0, stream>>>(x_raw, in_w, in_b, x);

    // ---- transformer layers ----
    for (int l = 0; l < NLAYER; l++) {
        ln_bf16_kernel<<<NTOK, HDIM, 0, stream>>>(x, ln1_s + l * HDIM, ln1_b + l * HDIM, h_bf);
        gemm(h_bf, wqkv_bf + (size_t)l * QKVDIM * HDIM, qkv_b + l * QKVDIM,
             nullptr, nullptr, qkv_bf, NTOK, HDIM, QKVDIM, 0);
        attn_wmma<<<(NHEAD * (NTOK / 16)) / 8, 256, 0, stream>>>(qkv_bf, mbits, o_bf);
        gemm(o_bf, wao_bf + (size_t)l * HDIM * HDIM, ao_b + l * HDIM,
             x, x, nullptr, NTOK, HDIM, HDIM, 0);                  // x += o @ ao_w^T + b
        ln_bf16_kernel<<<NTOK, HDIM, 0, stream>>>(x, ln2_s + l * HDIM, ln2_b + l * HDIM, h_bf);
        gemm(h_bf, wf1_bf + (size_t)l * FFDIM * HDIM, f1_b + l * FFDIM,
             nullptr, nullptr, ff_bf, NTOK, HDIM, FFDIM, 1);       // relu
        gemm(ff_bf, wf2_bf + (size_t)l * HDIM * FFDIM, f2_b + l * HDIM,
             x, x, nullptr, NTOK, FFDIM, HDIM, 0);                 // x += ffn
    }

    // ---- heads ----
    cvt(x, xb_bf, NTOK * HDIM);
    gemm(xb_bf, wl1_bf, lat1_b, nullptr, nullptr, t_bf, NTOK, HDIM, HDIM, 1);
    gemm(t_bf, wl2_bf, lat2_b, nullptr, out_lat, nullptr, NTOK, HDIM, 16, 0);
    latnorm_kernel<<<(NTOK + 255) / 256, 256, 0, stream>>>(out_lat);
    gemm(xb_bf, wb1_bf, b1_b, nullptr, nullptr, bb_bf, NTOK, HDIM, 64, 1);
    beta_kernel<<<(NTOK + 255) / 256, 256, 0, stream>>>(bb_bf, b2_w, b2_b, out_beta);
}
